// GroupedQueryAttention_52201032515730
// MI455X (gfx1250) — matrix operand order repaired and checked by reading, not compiled, not run on hardware
//
#include <hip/hip_runtime.h>
#include <hip/hip_bf16.h>

// ---------------------------------------------------------------------------
// GQA attention layer for gfx1250 (MI455X): bf16 WMMA + async global->LDS.
//   B=4, S=2048, D=1024, H=16, Hkv=4, d=64, soft-cap 30, causal, RMSNorm+RoPE.
// ---------------------------------------------------------------------------

#define B_N      4
#define S_LEN    2048
#define D_MODEL  1024
#define N_HEADS  16
#define N_KV     4
#define HEAD_DIM 64
#define KV_DIM   (N_KV * HEAD_DIM)     // 256
#define SCALE_F  0.125f                // 64^-0.5
#define CAP_F    30.0f

typedef unsigned short u16;
typedef __attribute__((ext_vector_type(16))) __bf16          v16bf;
typedef __attribute__((ext_vector_type(8)))  float           v8f;
typedef __attribute__((ext_vector_type(8)))  unsigned short  u16x8;

union Frag { v16bf v; u16x8 h[2]; };

__device__ __forceinline__ u16 f2bf(float f) {
  unsigned u = __float_as_uint(f);
  unsigned r = (u + 0x7FFFu + ((u >> 16) & 1u)) >> 16;   // round-to-nearest-even
  return (u16)r;
}

__device__ __forceinline__ v8f wmma_bf16(const Frag& a, const Frag& b, v8f c) {
  return __builtin_amdgcn_wmma_f32_16x16x32_bf16(
      /*neg_a=*/false, a.v, /*neg_b=*/false, b.v,
      /*c_mod=*/(short)0, c, /*reuse_a=*/false, /*reuse_b=*/false);
}

// generic (flat) shared pointer -> 0-based LDS byte offset (addrspace(3) value)
__device__ __forceinline__ unsigned lds_off(const void* p) {
  return (unsigned)(uintptr_t)(const __attribute__((address_space(3))) void*)p;
}

// CDNA5 async copy: GLOBAL_LOAD_ASYNC_TO_LDS_B128 (tracked by ASYNCcnt).
// VDST = per-lane LDS byte address, VADDR = per-lane 64-bit global address.
__device__ __forceinline__ void async_ld_b128(u16* ldsDst, const u16* g) {
  unsigned lo = lds_off(ldsDst);
  asm volatile("global_load_async_to_lds_b128 %0, %1, off"
               :: "v"(lo), "v"(g) : "memory");
}
__device__ __forceinline__ void wait_async0() {
  asm volatile("s_wait_asynccnt 0x0" ::: "memory");
}

// ---------------------------------------------------------------------------
// 1) f32 -> bf16 conversion
// ---------------------------------------------------------------------------
__global__ __launch_bounds__(256)
void cvt_f32_to_bf16(const float* __restrict__ s, u16* __restrict__ d, size_t n) {
  size_t i = (size_t)blockIdx.x * 256 + threadIdx.x;
  if (i < n) d[i] = f2bf(s[i]);
}

// ---------------------------------------------------------------------------
// 2) Y[M,N] = A[M,K](bf16) @ W[N,K]^T (bf16), f32 out.
//    Block tile 64x128, 8 waves (2x4), wave tile 32x32 => 4 WMMA / k-step.
//    Double-buffered LDS staged with async global->LDS b128; tile t+1 issued
//    before the WMMA work of tile t (HBM latency hidden behind matrix ops).
// ---------------------------------------------------------------------------
#define LSTR 40   // LDS row stride (u16): 80 B, 16B-aligned, bank-staggered
__device__ __forceinline__ void gemm_stage_async(
    const u16* __restrict__ A, const u16* __restrict__ W,
    u16* laBuf, u16* lwBuf, int m0, int n0, int K, int k0, int tid) {
  { // A tile 64x32: one b128 per thread
    int r = tid >> 2, c = (tid & 3) * 8;
    async_ld_b128(&laBuf[r * LSTR + c], &A[(size_t)(m0 + r) * K + k0 + c]);
  }
  { // W tile 128x32: two b128 per thread
    int r = tid >> 1, c = (tid & 1) * 16;
    const u16* wp = &W[(size_t)(n0 + r) * K + k0 + c];
    async_ld_b128(&lwBuf[r * LSTR + c], wp);
    async_ld_b128(&lwBuf[r * LSTR + c + 8], wp + 8);
  }
}

__global__ __launch_bounds__(256)
void gemm_bf16(const u16* __restrict__ A, const u16* __restrict__ W,
               float* __restrict__ Y, int M, int N, int K) {
  __shared__ u16 la[2][64 * LSTR];
  __shared__ u16 lw[2][128 * LSTR];
  const int tid  = threadIdx.x;
  const int lane = tid & 31, wave = tid >> 5;
  const int wm = wave >> 2;          // 0..1 (M dir)
  const int wn = wave & 3;           // 0..3 (N dir)
  const int m0 = blockIdx.x * 64;
  const int n0 = blockIdx.y * 128;

  v8f acc[2][2] = {};

  gemm_stage_async(A, W, la[0], lw[0], m0, n0, K, 0, tid);
  wait_async0();
  __syncthreads();

  int buf = 0;
  for (int k0 = 0; k0 < K; k0 += 32) {
    if (k0 + 32 < K)   // prefetch next slab straight into the other LDS buffer
      gemm_stage_async(A, W, la[buf ^ 1], lw[buf ^ 1], m0, n0, K, k0 + 32, tid);

    Frag a[2], bf[2];
#pragma unroll
    for (int i = 0; i < 2; i++) { // A frag: lane-half holds K 0-15 or 16-31
      const u16* p = &la[buf][(wm * 32 + i * 16 + (lane & 15)) * LSTR + (lane >> 4) * 16];
      a[i].h[0] = *(const u16x8*)p;
      a[i].h[1] = *(const u16x8*)(p + 8);
    }
#pragma unroll
    for (int j = 0; j < 2; j++) { // B frag: lane = column N, 16 contiguous K
      const u16* p = &lw[buf][(wn * 32 + j * 16 + (lane & 15)) * LSTR + (lane >> 4) * 16];
      bf[j].h[0] = *(const u16x8*)p;
      bf[j].h[1] = *(const u16x8*)(p + 8);
    }
#pragma unroll
    for (int i = 0; i < 2; i++)
#pragma unroll
      for (int j = 0; j < 2; j++)
        acc[i][j] = wmma_bf16(a[i], bf[j], acc[i][j]);

    wait_async0();        // next buffer's async writes have landed
    __syncthreads();      // everyone done reading current buffer
    buf ^= 1;
  }

  // C layout: VGPR r -> row r (lanes 0-15) / row 8+r (lanes 16-31), col = lane&15
#pragma unroll
  for (int i = 0; i < 2; i++)
#pragma unroll
    for (int j = 0; j < 2; j++) {
      int rb = m0 + wm * 32 + i * 16 + ((lane >> 4) << 3);
      int cc = n0 + wn * 32 + j * 16 + (lane & 15);
#pragma unroll
      for (int r = 0; r < 8; r++)
        Y[(size_t)(rb + r) * N + cc] = acc[i][j][r];
    }
}

// ---------------------------------------------------------------------------
// 3) Per-(token,head): RMSNorm(64) + RoPE, relayout to [B,Hh,S,64] bf16.
//    One wave32 per task; lane handles elems {lane, lane+32} (RoPE pairing).
// ---------------------------------------------------------------------------
__global__ __launch_bounds__(256)
void rmsnorm_rope_relayout(const float* __restrict__ src, int Hh,
                           const float* __restrict__ nw, int do_norm_rope,
                           u16* __restrict__ dst) {
  const int lane = threadIdx.x & 31, wave = threadIdx.x >> 5;
  const long id = (long)blockIdx.x * 8 + wave;
  const int head = (int)(id % Hh);
  const long token = id / Hh;
  const int b = (int)(token / S_LEN), s = (int)(token % S_LEN);

  const float* p = src + (size_t)token * (Hh * HEAD_DIM) + head * HEAD_DIM;
  float f0 = p[lane], f1 = p[lane + 32];
  u16 o0, o1;
  if (do_norm_rope) {
    float ss = f0 * f0 + f1 * f1;
#pragma unroll
    for (int m = 16; m >= 1; m >>= 1) ss += __shfl_xor(ss, m, 32);
    float r = rsqrtf(ss * (1.0f / 64.0f) + 1e-6f);
    float q0 = f0 * r * nw[lane];
    float q1 = f1 * r * nw[lane + 32];
    // inv_freq = theta^(-2*lane/64); ln(10000) = 9.210340371976184
    float ang = (float)s * __expf((float)lane * (-2.0f / 64.0f) * 9.2103403720f);
    float sn, cs;
    __sincosf(ang, &sn, &cs);
    o0 = f2bf(q0 * cs - q1 * sn);
    o1 = f2bf(q1 * cs + q0 * sn);
  } else {
    o0 = f2bf(f0);
    o1 = f2bf(f1);
  }
  u16* q = dst + ((size_t)(b * Hh + head) * S_LEN + s) * HEAD_DIM;
  q[lane] = o0;
  q[lane + 32] = o1;
}

// ---------------------------------------------------------------------------
// 4) Flash attention, causal, tanh soft-cap. Block = (b, h, 64 q-rows),
//    128 threads = 4 waves; each wave owns 16 q-rows. Streams 32-key tiles:
//    K tile staged with async global->LDS; scores: 4 WMMA, PV: 4 WMMA.
// ---------------------------------------------------------------------------
#define KSTR 72   // K tile row stride (u16), 144 B
#define VSTR 40   // transposed V tile / P tile row stride (u16), 80 B
__global__ __launch_bounds__(128)
void flash_attn(const u16* __restrict__ Qb, const u16* __restrict__ Kb,
                const u16* __restrict__ Vb, u16* __restrict__ Ob) {
  __shared__ u16 ks[32 * KSTR];           // K tile: [key][d]
  __shared__ u16 vs[64 * VSTR];           // V tile transposed: [d][key]
  __shared__ u16 ps[4 * 16 * VSTR];       // per-wave P tile: [row][key]

  const int tid = threadIdx.x, lane = tid & 31, wave = tid >> 5;
  const int qt = blockIdx.x, h = blockIdx.y, b = blockIdx.z;
  const int hkv = h >> 2;                 // N_REP = 4
  const u16* Qh = Qb + (size_t)(b * N_HEADS + h) * S_LEN * HEAD_DIM;
  const u16* Kh = Kb + (size_t)(b * N_KV + hkv) * S_LEN * HEAD_DIM;
  const u16* Vh = Vb + (size_t)(b * N_KV + hkv) * S_LEN * HEAD_DIM;
  const int qrow0 = qt * 64 + wave * 16;

  // Q fragments (16 rows x 64 d) held in registers for the whole pass
  Frag qf[2];
#pragma unroll
  for (int dc = 0; dc < 2; dc++) {
    const u16* qp = Qh + (size_t)(qrow0 + (lane & 15)) * HEAD_DIM
                       + dc * 32 + (lane >> 4) * 16;
    qf[dc].h[0] = *(const u16x8*)qp;
    qf[dc].h[1] = *(const u16x8*)(qp + 8);
  }

  v8f o[4] = {};
  float l[8], m[8];
#pragma unroll
  for (int r = 0; r < 8; r++) { l[r] = 0.0f; m[r] = -3.0e38f; }

  const int ntiles = qt * 2 + 2;          // keys 0 .. qt*64+63
  for (int t = 0; t < ntiles; t++) {
    const int kb = t * 32;
    { // stage K tile 32x64 row-major: async global->LDS, 2 x b128 per thread
      int r = tid >> 2, c = (tid & 3) * 16;
      const u16* kp = Kh + (size_t)(kb + r) * HEAD_DIM + c;
      async_ld_b128(&ks[r * KSTR + c], kp);
      async_ld_b128(&ks[r * KSTR + c + 8], kp + 8);
    }
    { // stage V tile transposed: vs[d][key] (manual transpose scatter)
      int r = tid >> 2, c = (tid & 3) * 16;
      const u16* vp = Vh + (size_t)(kb + r) * HEAD_DIM + c;
      u16x8 t0 = *(const u16x8*)vp;
      u16x8 t1 = *(const u16x8*)(vp + 8);
#pragma unroll
      for (int e = 0; e < 8; e++) vs[(c + e) * VSTR + r] = t0[e];
#pragma unroll
      for (int e = 0; e < 8; e++) vs[(c + 8 + e) * VSTR + r] = t1[e];
    }
    wait_async0();
    __syncthreads();

    // ---- scores S = Q K^T : two key groups x two d-chunks => 4 WMMA ----
    v8f c0 = {}, c1 = {};
#pragma unroll
    for (int dc = 0; dc < 2; dc++) {
      Frag bf;
      const u16* p0 = &ks[(lane & 15) * KSTR + dc * 32 + (lane >> 4) * 16];
      bf.h[0] = *(const u16x8*)p0;
      bf.h[1] = *(const u16x8*)(p0 + 8);
      c0 = wmma_bf16(qf[dc], bf, c0);
      const u16* p1 = &ks[(16 + (lane & 15)) * KSTR + dc * 32 + (lane >> 4) * 16];
      bf.h[0] = *(const u16x8*)p1;
      bf.h[1] = *(const u16x8*)(p1 + 8);
      c1 = wmma_bf16(qf[dc], bf, c1);
    }

    // ---- soft-cap + causal mask + online softmax (f32 VALU) ----
    float p0[8], p1[8], sf[8];
#pragma unroll
    for (int r = 0; r < 8; r++) {
      int qpos = qrow0 + r + ((lane >> 4) << 3);
      float s0 = CAP_F * tanhf(c0[r] * SCALE_F * (1.0f / CAP_F));
      float s1 = CAP_F * tanhf(c1[r] * SCALE_F * (1.0f / CAP_F));
      int k0 = kb + (lane & 15);
      if (k0 > qpos)      s0 = -1e30f;
      if (k0 + 16 > qpos) s1 = -1e30f;
      float mx = fmaxf(s0, s1);          // row max across 16 lanes of this half
      mx = fmaxf(mx, __shfl_xor(mx, 1, 32));
      mx = fmaxf(mx, __shfl_xor(mx, 2, 32));
      mx = fmaxf(mx, __shfl_xor(mx, 4, 32));
      mx = fmaxf(mx, __shfl_xor(mx, 8, 32));
      float mn = fmaxf(m[r], mx);
      sf[r] = __expf(m[r] - mn);
      p0[r] = __expf(s0 - mn);
      p1[r] = __expf(s1 - mn);
      float rs = p0[r] + p1[r];
      rs += __shfl_xor(rs, 1, 32);
      rs += __shfl_xor(rs, 2, 32);
      rs += __shfl_xor(rs, 4, 32);
      rs += __shfl_xor(rs, 8, 32);
      l[r] = l[r] * sf[r] + rs;
      m[r] = mn;
    }
#pragma unroll
    for (int tt = 0; tt < 4; tt++)
#pragma unroll
      for (int r = 0; r < 8; r++) o[tt][r] *= sf[r];

    // ---- P: C-layout -> LDS -> A-layout fragment ----
    {
      u16* pw = &ps[wave * 16 * VSTR];
      int rr = (lane >> 4) << 3;
#pragma unroll
      for (int r = 0; r < 8; r++) {
        pw[(r + rr) * VSTR + (lane & 15)]      = f2bf(p0[r]);
        pw[(r + rr) * VSTR + 16 + (lane & 15)] = f2bf(p1[r]);
      }
    }
    Frag pf;
    {
      const u16* pr = &ps[wave * 16 * VSTR + (lane & 15) * VSTR + (lane >> 4) * 16];
      pf.h[0] = *(const u16x8*)pr;
      pf.h[1] = *(const u16x8*)(pr + 8);
    }

    // ---- O += P V : four 16-wide d tiles => 4 WMMA ----
#pragma unroll
    for (int tt = 0; tt < 4; tt++) {
      Frag bf;
      const u16* vr = &vs[(tt * 16 + (lane & 15)) * VSTR + (lane >> 4) * 16];
      bf.h[0] = *(const u16x8*)vr;
      bf.h[1] = *(const u16x8*)(vr + 8);
      o[tt] = wmma_bf16(pf, bf, o[tt]);
    }
    __syncthreads();
  }

  // normalize + store bf16 in [token][h*64+d] layout (A operand of final GEMM)
#pragma unroll
  for (int tt = 0; tt < 4; tt++)
#pragma unroll
    for (int r = 0; r < 8; r++) {
      int qrow = qrow0 + r + ((lane >> 4) << 3);
      int col = h * HEAD_DIM + tt * 16 + (lane & 15);
      Ob[(size_t)(b * S_LEN + qrow) * D_MODEL + col] = f2bf(o[tt][r] / l[r]);
    }
}

// ---------------------------------------------------------------------------
// Launcher
// ---------------------------------------------------------------------------
extern "C" void kernel_launch(void* const* d_in, const int* in_sizes, int n_in,
                              void* d_out, int out_size, void* d_ws, size_t ws_size,
                              hipStream_t stream) {
  const float* x  = (const float*)d_in[0];
  const float* Wq = (const float*)d_in[1];
  const float* Wk = (const float*)d_in[2];
  const float* Wv = (const float*)d_in[3];
  const float* Wo = (const float*)d_in[4];
  const float* qn = (const float*)d_in[5];
  const float* kn = (const float*)d_in[6];
  float* out = (float*)d_out;
  (void)in_sizes; (void)n_in; (void)out_size; (void)ws_size;

  const size_t TOK = (size_t)B_N * S_LEN;   // 8192
  size_t off = 0;
  auto walloc = [&](size_t bytes) -> void* {
    void* p = (char*)d_ws + off;
    off += (bytes + 255) & ~(size_t)255;
    return p;
  };
  u16*   Xb  = (u16*)walloc(TOK * D_MODEL * 2);
  u16*   Wqb = (u16*)walloc((size_t)D_MODEL * D_MODEL * 2);
  u16*   Wkb = (u16*)walloc((size_t)KV_DIM * D_MODEL * 2);
  u16*   Wvb = (u16*)walloc((size_t)KV_DIM * D_MODEL * 2);
  u16*   Wob = (u16*)walloc((size_t)D_MODEL * D_MODEL * 2);
  float* Qf  = (float*)walloc(TOK * D_MODEL * 4);
  float* Kf  = (float*)walloc(TOK * KV_DIM * 4);
  float* Vf  = (float*)walloc(TOK * KV_DIM * 4);
  u16*   Qbf = (u16*)walloc(TOK * D_MODEL * 2);   // [B,H,S,64]
  u16*   Kbf = (u16*)walloc(TOK * KV_DIM * 2);    // [B,Hkv,S,64]
  u16*   Vbf = (u16*)walloc(TOK * KV_DIM * 2);
  u16*   Ob  = (u16*)walloc(TOK * D_MODEL * 2);   // [token][H*64]

  auto cvt = [&](const float* s, u16* d, size_t n) {
    cvt_f32_to_bf16<<<(unsigned)((n + 255) / 256), 256, 0, stream>>>(s, d, n);
  };
  cvt(x,  Xb,  TOK * D_MODEL);
  cvt(Wq, Wqb, (size_t)D_MODEL * D_MODEL);
  cvt(Wk, Wkb, (size_t)KV_DIM * D_MODEL);
  cvt(Wv, Wvb, (size_t)KV_DIM * D_MODEL);
  cvt(Wo, Wob, (size_t)D_MODEL * D_MODEL);

  dim3 blk(256);
  gemm_bf16<<<dim3(TOK / 64, D_MODEL / 128), blk, 0, stream>>>(
      Xb, Wqb, Qf, (int)TOK, D_MODEL, D_MODEL);
  gemm_bf16<<<dim3(TOK / 64, KV_DIM / 128), blk, 0, stream>>>(
      Xb, Wkb, Kf, (int)TOK, KV_DIM, D_MODEL);
  gemm_bf16<<<dim3(TOK / 64, KV_DIM / 128), blk, 0, stream>>>(
      Xb, Wvb, Vf, (int)TOK, KV_DIM, D_MODEL);

  rmsnorm_rope_relayout<<<(unsigned)(TOK * N_HEADS / 8), 256, 0, stream>>>(
      Qf, N_HEADS, qn, 1, Qbf);
  rmsnorm_rope_relayout<<<(unsigned)(TOK * N_KV / 8), 256, 0, stream>>>(
      Kf, N_KV, kn, 1, Kbf);
  rmsnorm_rope_relayout<<<(unsigned)(TOK * N_KV / 8), 256, 0, stream>>>(
      Vf, N_KV, (const float*)nullptr, 0, Vbf);

  flash_attn<<<dim3(S_LEN / 64, N_HEADS, B_N), dim3(128), 0, stream>>>(
      Qbf, Kbf, Vbf, Ob);

  gemm_bf16<<<dim3(TOK / 64, D_MODEL / 128), blk, 0, stream>>>(
      Ob, Wob, out, (int)TOK, D_MODEL, D_MODEL);
}
